// FeedBack_57260503991041
// MI455X (gfx1250) — matrix operand order repaired and checked by reading, not compiled, not run on hardware
//
#include <hip/hip_runtime.h>
#include <hip/hip_bf16.h>
#include <math.h>

// Problem constants (match reference)
#define BQ 512
#define TQ 128
#define FQ 64
#define UQ 1024
#define OUT_STEPS 24
#define NG (4 * UQ)          // 4096 gate columns
#define KTOT (FQ + UQ)       // 1088 packed K (x | h)

typedef __attribute__((ext_vector_type(16))) _Float16 v16h;
typedef __attribute__((ext_vector_type(8)))  _Float16 v8h;
typedef __attribute__((ext_vector_type(8)))  float    v8f;

static __device__ __forceinline__ v16h cat8(v8h lo, v8h hi) {
    return __builtin_shufflevector(lo, hi, 0,1,2,3,4,5,6,7,8,9,10,11,12,13,14,15);
}

// CDNA5 async global->LDS copy (16 bytes per lane, tracked by ASYNCcnt).
// Low 32 bits of a flat shared pointer == wave-relative LDS offset.
static __device__ __forceinline__ void async_b128(void* lds, const void* gm) {
    asm volatile("global_load_async_to_lds_b128 %0, %1, off"
                 :: "v"((unsigned)(unsigned long long)lds), "v"(gm)
                 : "memory");
}
static __device__ __forceinline__ void wait_async0() {
    asm volatile("s_wait_asynccnt 0x0" ::: "memory");
}

// ---------------------------------------------------------------------------
// Pack Wx (64x4096) and Wh (1024x4096) fp32 -> f16, transposed to [n][k],
// n in [0,4096), k in [0,1088): k<64 -> Wx row, else Wh row.
// ---------------------------------------------------------------------------
__global__ void k_pack_w(const float* __restrict__ Wx, const float* __restrict__ Wh,
                         _Float16* __restrict__ W16) {
    int idx = blockIdx.x * 256 + threadIdx.x;
    if (idx >= NG * KTOT) return;
    int n = idx / KTOT;
    int k = idx - n * KTOT;
    float v = (k < FQ) ? Wx[k * NG + n] : Wh[(k - FQ) * NG + n];
    W16[idx] = (_Float16)v;
}

// Pack Wd (1024x64) fp32 -> f16 transposed to [n][k], n<64, k<1024.
__global__ void k_pack_wd(const float* __restrict__ Wd, _Float16* __restrict__ Wd16) {
    int idx = blockIdx.x * 256 + threadIdx.x;
    if (idx >= FQ * UQ) return;
    int n = idx >> 10;
    int k = idx & 1023;
    Wd16[idx] = (_Float16)Wd[k * FQ + n];
}

// Zero both A ping-pong buffers (f16, as u32 words) and the c state (fp32).
__global__ void k_init(unsigned* __restrict__ A0, unsigned* __restrict__ A1,
                       float* __restrict__ c) {
    int idx = blockIdx.x * 256 + threadIdx.x;
    if (idx < (BQ * KTOT) / 2) { A0[idx] = 0u; A1[idx] = 0u; }
    if (idx < BQ * UQ) c[idx] = 0.0f;
}

// Load x_t (fp32 [B,T,F] slice) into the x region (cols 0..63) of A buffer.
__global__ void k_load_x(const float* __restrict__ inputs, int t,
                         _Float16* __restrict__ A) {
    int idx = blockIdx.x * 256 + threadIdx.x;   // < 512*64
    int m = idx >> 6;
    int f = idx & 63;
    A[m * KTOT + f] = (_Float16)inputs[(m * TQ + t) * FQ + f];
}

// ---------------------------------------------------------------------------
// One fused LSTM step.
//   z[g] = A @ W[:, g*U + N0 : ...]  (A = [x|h] f16 512x1088, W pre-transposed)
//   i,f,g,o -> c_new, h_new;  h_new (f16) written into Anxt's h region.
// Block: 256 threads (8 waves). Tile: 64(M) x 64(U) x 4 gates.
// Wave (wm2, wn): wm2 in 0..1 -> two M subtiles (B-frag reuse), wn in 0..3 ->
// one 16-col N subtile. Double-buffered LDS fed by async global->LDS copies.
// K loop fully unrolled so the ping-pong index and LDS offsets are constants
// (avoids accumulator phi copies / v_nop hazards).
// ---------------------------------------------------------------------------
__global__ __launch_bounds__(256) void k_step(
    const _Float16* __restrict__ W16,   // [4096][1088]
    const _Float16* __restrict__ Acur,  // [512][1088]
    _Float16* __restrict__ Anxt,        // [512][1088] (h region written)
    float* __restrict__ c,              // [512][1024]
    const float* __restrict__ bias)     // [4096]
{
    __shared__ __align__(16) _Float16 Als[2][64][32];    //  2 x 4 KB
    __shared__ __align__(16) _Float16 Bls[2][256][32];   //  2 x 16 KB

    const int tid  = threadIdx.x;
    const int M0   = blockIdx.x * 64;
    const int N0   = blockIdx.y * 64;
    const int wave = tid >> 5;
    const int lane = tid & 31;
    const int wm2  = wave >> 2;          // 0..1 : M subtile pair
    const int wn   = wave & 3;           // 0..3 : N subtile
    const int lhi  = lane >> 4;
    const int llo  = lane & 15;

    // Per-thread staging assignment (same every iteration)
    const int sam = tid >> 2;            // A: row 0..63
    const int sak = (tid & 3) * 8;       // A: k-offset 0/8/16/24
    const int sg  = tid >> 6;            // B: gate 0..3
    const int sj  = tid & 63;            // B: column within tile
    const _Float16* srcA = &Acur[(M0 + sam) * KTOT + sak];
    const _Float16* srcB = &W16[(((sg << 10) + N0 + sj)) * KTOT];

    v8f acc[4][2] = {};   // [gate][m-subtile]

    // Prologue: async-stage k0 = 0 into buffer 0
    async_b128(&Als[0][sam][sak], srcA);
    async_b128(&Bls[0][tid][0],  srcB + 0);
    async_b128(&Bls[0][tid][8],  srcB + 8);
    async_b128(&Bls[0][tid][16], srcB + 16);
    async_b128(&Bls[0][tid][24], srcB + 24);

#pragma unroll
    for (int kk = 0; kk < KTOT / 32; ++kk) {
        const int p = kk & 1;                    // compile-time after unroll
        wait_async0();
        __syncthreads();

        const int knext = (kk + 1) * 32;
        if (knext < KTOT) {                      // folds away after unroll
            const int pn = p ^ 1;
            async_b128(&Als[pn][sam][sak], srcA + knext);
            async_b128(&Bls[pn][tid][0],  srcB + knext + 0);
            async_b128(&Bls[pn][tid][8],  srcB + knext + 8);
            async_b128(&Bls[pn][tid][16], srcB + knext + 16);
            async_b128(&Bls[pn][tid][24], srcB + knext + 24);
        }

        // A fragments for the two M subtiles (CDNA5 16-bit A layout)
        const int am = wm2 * 32 + llo;
        const int kb = lhi * 8;
        v16h a0 = cat8(*(const v8h*)&Als[p][am][kb],
                       *(const v8h*)&Als[p][am][16 + kb]);
        v16h a1 = cat8(*(const v8h*)&Als[p][am + 16][kb],
                       *(const v8h*)&Als[p][am + 16][16 + kb]);
        const int kbb = lhi * 8;
#pragma unroll
        for (int g = 0; g < 4; ++g) {
            int row = g * 64 + wn * 16 + llo;
            v16h bf = cat8(*(const v8h*)&Bls[p][row][kbb],
                           *(const v8h*)&Bls[p][row][kbb + 16]);
            acc[g][0] = __builtin_amdgcn_wmma_f32_16x16x32_f16(
                false, a0, false, bf, (short)0, acc[g][0], false, false);
            acc[g][1] = __builtin_amdgcn_wmma_f32_16x16x32_f16(
                false, a1, false, bf, (short)0, acc[g][1], false, false);
        }
    }

    // Fused LSTM cell epilogue
    const int n_u = N0 + wn * 16 + llo;              // 0..1023
    const float bi = bias[n_u];
    const float bf_ = bias[UQ + n_u];
    const float bg = bias[2 * UQ + n_u];
    const float bo = bias[3 * UQ + n_u];
#pragma unroll
    for (int ms = 0; ms < 2; ++ms) {
#pragma unroll
        for (int v = 0; v < 8; ++v) {
            int m = M0 + wm2 * 32 + ms * 16 + lhi * 8 + v;  // C/D: M = v + 8*(lane>=16)
            float zi = acc[0][ms][v] + bi;
            float zf = acc[1][ms][v] + bf_;
            float zg = acc[2][ms][v] + bg;
            float zo = acc[3][ms][v] + bo;
            float ig = 1.0f / (1.0f + __expf(-zi));
            float fg = 1.0f / (1.0f + __expf(-zf));
            float gg = tanhf(zg);
            float og = 1.0f / (1.0f + __expf(-zo));
            int ci = m * UQ + n_u;
            float cn = fg * c[ci] + ig * gg;
            c[ci] = cn;
            Anxt[m * KTOT + FQ + n_u] = (_Float16)(og * tanhf(cn));
        }
    }
}

// ---------------------------------------------------------------------------
// Readout: p = h @ Wd + bd.  Reads h region of A, writes fp32 prediction to
// out[:, step, :] and f16 prediction into A's x region (feedback input).
// Block: 256 threads, tile 64(M) x 64(N full), K = 1024.
// ---------------------------------------------------------------------------
__global__ __launch_bounds__(256) void k_readout(
    const _Float16* __restrict__ Wd16,  // [64][1024]
    _Float16* __restrict__ A,           // [512][1088]
    const float* __restrict__ bd,       // [64]
    float* __restrict__ out,            // [512][24][64]
    int step)
{
    __shared__ __align__(16) _Float16 Als[64][32];
    __shared__ __align__(16) _Float16 Bls[64][32];

    const int tid  = threadIdx.x;
    const int M0   = blockIdx.x * 64;
    const int wave = tid >> 5;
    const int lane = tid & 31;
    const int wm   = wave >> 1;
    const int wn   = wave & 1;
    const int lhi  = lane >> 4;
    const int llo  = lane & 15;

    v8f acc[2] = {};

#pragma unroll
    for (int kk = 0; kk < UQ / 32; ++kk) {
        const int k0 = kk * 32;
        {
            int m  = tid >> 2;
            int kq = (tid & 3) * 8;
            *(v8h*)&Als[m][kq] = *(const v8h*)&A[(M0 + m) * KTOT + FQ + k0 + kq];
        }
        {
            int col = tid >> 2;
            int kq  = (tid & 3) * 8;
            *(v8h*)&Bls[col][kq] = *(const v8h*)&Wd16[col * UQ + k0 + kq];
        }
        __syncthreads();

        const int am = wm * 16 + llo;
        const int kb = lhi * 8;
        v16h afrag = cat8(*(const v8h*)&Als[am][kb],
                          *(const v8h*)&Als[am][16 + kb]);
        const int kbb = lhi * 8;
#pragma unroll
        for (int s = 0; s < 2; ++s) {
            int row = wn * 32 + s * 16 + llo;
            v16h bfrag = cat8(*(const v8h*)&Bls[row][kbb],
                              *(const v8h*)&Bls[row][kbb + 16]);
            acc[s] = __builtin_amdgcn_wmma_f32_16x16x32_f16(
                false, afrag, false, bfrag, (short)0, acc[s], false, false);
        }
        __syncthreads();
    }

#pragma unroll
    for (int s = 0; s < 2; ++s) {
        int n = wn * 32 + s * 16 + llo;               // 0..63
        float bb = bd[n];
#pragma unroll
        for (int v = 0; v < 8; ++v) {
            int m = M0 + wm * 16 + lhi * 8 + v;
            float z = acc[s][v] + bb;
            out[(m * OUT_STEPS + step) * FQ + n] = z;
            A[m * KTOT + n] = (_Float16)z;            // feedback x for next step
        }
    }
}

// ---------------------------------------------------------------------------
extern "C" void kernel_launch(void* const* d_in, const int* in_sizes, int n_in,
                              void* d_out, int out_size, void* d_ws, size_t ws_size,
                              hipStream_t stream) {
    const float* inputs = (const float*)d_in[0];   // [512,128,64]
    const float* Wx     = (const float*)d_in[1];   // [64,4096]
    const float* Wh     = (const float*)d_in[2];   // [1024,4096]
    const float* bias   = (const float*)d_in[3];   // [4096]
    const float* Wd     = (const float*)d_in[4];   // [1024,64]
    const float* bd     = (const float*)d_in[5];   // [64]
    float* out = (float*)d_out;                    // [512,24,64]

    // Workspace layout (bytes, 256-aligned)
    char* ws = (char*)d_ws;
    _Float16* W16  = (_Float16*)(ws);                               // 4096*1088*2 = 8,912,896
    _Float16* Wd16 = (_Float16*)(ws + 8912896);                     // 64*1024*2   =   131,072
    _Float16* A0   = (_Float16*)(ws + 9043968);                     // 512*1088*2  = 1,114,112
    _Float16* A1   = (_Float16*)(ws + 10158080);                    // 512*1088*2  = 1,114,112
    float*    cbuf = (float*)   (ws + 11272192);                    // 512*1024*4  = 2,097,152
    _Float16* Abuf[2] = { A0, A1 };

    // One-time-per-launch packing + state init
    k_pack_w <<<(NG * KTOT + 255) / 256, 256, 0, stream>>>(Wx, Wh, W16);
    k_pack_wd<<<(FQ * UQ + 255) / 256,   256, 0, stream>>>(Wd, Wd16);
    k_init   <<<(BQ * UQ + 255) / 256,   256, 0, stream>>>((unsigned*)A0, (unsigned*)A1, cbuf);

    dim3 gstep(BQ / 64, UQ / 64);   // 8 x 16 blocks
    int cur = 0;

    // Warmup over the 128-step input window
    for (int t = 0; t < TQ; ++t) {
        k_load_x<<<(BQ * FQ) / 256, 256, 0, stream>>>(inputs, t, Abuf[cur]);
        k_step  <<<gstep, 256, 0, stream>>>(W16, Abuf[cur], Abuf[cur ^ 1], cbuf, bias);
        cur ^= 1;
    }

    // First prediction from the final warmup state
    k_readout<<<BQ / 64, 256, 0, stream>>>(Wd16, Abuf[cur], bd, out, 0);

    // Autoregressive decode
    for (int s = 1; s < OUT_STEPS; ++s) {
        k_step  <<<gstep, 256, 0, stream>>>(W16, Abuf[cur], Abuf[cur ^ 1], cbuf, bias);
        cur ^= 1;
        k_readout<<<BQ / 64, 256, 0, stream>>>(Wd16, Abuf[cur], bd, out, s);
    }
}
